// GNN_36232344109064
// MI455X (gfx1250) — matrix operand order repaired and checked by reading, not compiled, not run on hardware
//
#include <hip/hip_runtime.h>
#include <hip/hip_bf16.h>
#include <math.h>

typedef __attribute__((ext_vector_type(16))) _Float16 v16h;
typedef __attribute__((ext_vector_type(8)))  _Float16 v8h;
typedef __attribute__((ext_vector_type(8)))  float    v8f;

#define N_NODES 1024
#define BATCH   64
#define MT      64    // column tile per workgroup
#define XROWS   64    // padded channel rows in activation buffers

// ---------------- WMMA fragment helpers (wave32, v_wmma_f32_16x16x32_f16) ----

// A-matrix 16x32 f16 fragment from row-major f16 memory (stride in elements).
// ISA layout: lane l (h=l>>4, r=l&15): elems 0..7 = K(8h..8h+7), 8..15 = K(16+8h..23+8h)
__device__ inline v16h load_a_frag(const _Float16* p, int stride) {
    const int l = threadIdx.x & 31;
    const int h = l >> 4, r = l & 15;
    const _Float16* q = p + r * stride + 8 * h;
    v8h lo = *(const v8h*)(q);
    v8h hi = *(const v8h*)(q + 16);
    v16h a;
#pragma unroll
    for (int i = 0; i < 8; ++i) { a[i] = lo[i]; a[8 + i] = hi[i]; }
    return a;
}

// B-matrix 32x16 f16 fragment from LDS holding B transposed as [col][K].
// lane l: col = l&15, elems 0..7 = K(8h..8h+7), 8..15 = K(16+8h..23+8h)
__device__ inline v16h load_b_frag(const _Float16* p, int stride) {
    const int l = threadIdx.x & 31;
    const int h = l >> 4, c = l & 15;
    const _Float16* q = p + c * stride + 8 * h;
    v8h lo = *(const v8h*)(q);
    v8h hi = *(const v8h*)(q + 16);
    v16h bf;
#pragma unroll
    for (int i = 0; i < 8; ++i) { bf[i] = lo[i]; bf[8 + i] = hi[i]; }
    return bf;
}

// A-fragment built from f32 weight matrix W[c_out_real x ldw], zero padded.
__device__ inline v16h load_w_frag(const float* W, int ldw, int o0, int k0, int c_out_real) {
    const int l = threadIdx.x & 31;
    const int h = l >> 4, r = l & 15;
    const int o = o0 + r;
    v16h a;
#pragma unroll
    for (int e = 0; e < 16; ++e) {
        int K = k0 + 8 * h + (e < 8 ? e : 8 + e);   // e>=8 -> 16+(e-8)
        float v = (o < c_out_real && K < ldw) ? W[o * ldw + K] : 0.0f;
        a[e] = (_Float16)v;
    }
    return a;
}

__device__ inline v8f wmma_f32(v16h a, v16h b, v8f c) {
    return __builtin_amdgcn_wmma_f32_16x16x32_f16(false, a, false, b, (short)0, c,
                                                  false, false);
}

// ---------------- fused graph-conv layer ------------------------------------
// Y = X @ adj (adj recomputed tile-wise, symmetric), cat=[X;Y], Z = relu(W@cat+b)
template <int CIN_REAL, int CIN_PAD, int CCAT_PAD, int COUT_REAL, int COUT_PAD, bool RELU>
__global__ __launch_bounds__(256) void layer_kernel(
    const _Float16* __restrict__ Xin, const float* __restrict__ phi,
    const float* __restrict__ eta, const float* __restrict__ W,
    const float* __restrict__ bias, _Float16* __restrict__ Xout)
{
    constexpr int ADJ_S = 40;            // 32 + pad (rows stay 16B aligned)
    constexpr int CAT_S = CCAT_PAD + 8;  // 16B aligned row stride
    constexpr int T1 = (CIN_PAD / 16) * (MT / 16);
    constexpr int T2 = (COUT_PAD / 16) * (MT / 16);
    constexpr int NT1 = (T1 + 7) / 8;
    constexpr int NT2 = (T2 + 7) / 8;

    __shared__ float phiL[N_NODES];
    __shared__ float etaL[N_NODES];
    __shared__ __align__(32) _Float16 adjT[MT][ADJ_S];  // [m][n] chunk (adj symmetric)
    __shared__ __align__(32) _Float16 catT[MT][CAT_S];  // [m][c]

    const int tid = threadIdx.x;
    const int wid = tid >> 5;
    const int b   = blockIdx.y;
    const int m0  = blockIdx.x * MT;
    const size_t xbase = (size_t)b * XROWS * N_NODES;

    // stage phi/eta for this batch (8 KB LDS)
#pragma unroll
    for (int i = 0; i < N_NODES / 256; ++i) {
        int idx = tid + i * 256;
        phiL[idx] = phi[b * N_NODES + idx];
        etaL[idx] = eta[b * N_NODES + idx];
    }
    __syncthreads();

    // ---- GEMM1: Y[CIN_PAD x MT] = X @ adj, K over N in chunks of 32 ----
    v8f acc1[NT1];
#pragma unroll
    for (int i = 0; i < NT1; ++i)
#pragma unroll
        for (int q = 0; q < 8; ++q) acc1[i][q] = 0.0f;

    const int am  = tid >> 2;         // 0..63 : m within tile
    const int an0 = (tid & 3) * 8;    // n start for this thread
    const float pm = phiL[m0 + am];
    const float em = etaL[m0 + am];

    for (int k0 = 0; k0 < N_NODES; k0 += 32) {
        // adjT[m][n] = exp(-((phi_m-phi_n)^2 + (eta_m-eta_n)^2)), n in [k0,k0+32)
#pragma unroll
        for (int j = 0; j < 8; ++j) {
            int n = an0 + j;
            float dp = pm - phiL[k0 + n];
            float de = em - etaL[k0 + n];
            adjT[am][n] = (_Float16)__expf(-(dp * dp + de * de));
        }
        __syncthreads();
#pragma unroll
        for (int i = 0; i < NT1; ++i) {
            int t = wid + i * 8;
            if (t < T1) {
                int ct = t >> 2, mt = t & 3;
                v16h a  = load_a_frag(Xin + xbase + (size_t)(ct * 16) * N_NODES + k0,
                                      N_NODES);
                v16h bf = load_b_frag(&adjT[mt * 16][0], ADJ_S);
                acc1[i] = wmma_f32(a, bf, acc1[i]);
            }
        }
        __syncthreads();
    }

    // ---- build catT = [X ; Y] transposed in LDS ----
    {   // zero (covers padded channel columns)
        unsigned int* cz = (unsigned int*)&catT[0][0];
        const int nwords = MT * CAT_S / 2;
        for (int idx = tid; idx < nwords; idx += 256) cz[idx] = 0u;
    }
    __syncthreads();
    // X part: catT[m][c] = X[c][m0+m]
    for (int idx = tid; idx < MT * CIN_REAL; idx += 256) {
        int c = idx >> 6;          // / MT
        int m = idx & (MT - 1);
        catT[m][c] = Xin[xbase + (size_t)c * N_NODES + m0 + m];
    }
    // Y part: catT[m][CIN_REAL + c] = Y[c][m]   (f32 -> f16)
    {
        const int l = tid & 31;
        const int h = l >> 4, cn = l & 15;
#pragma unroll
        for (int i = 0; i < NT1; ++i) {
            int t = wid + i * 8;
            if (t < T1) {
                int ct = t >> 2, mt = t & 3;
#pragma unroll
                for (int r = 0; r < 8; ++r) {
                    int c = ct * 16 + r + 8 * h;
                    if (c < CIN_REAL)
                        catT[mt * 16 + cn][CIN_REAL + c] = (_Float16)acc1[i][r];
                }
            }
        }
    }
    __syncthreads();

    // ---- GEMM2: Z = W @ cat ----
    v8f acc2[NT2];
#pragma unroll
    for (int i = 0; i < NT2; ++i)
#pragma unroll
        for (int q = 0; q < 8; ++q) acc2[i][q] = 0.0f;

#pragma unroll
    for (int kc = 0; kc < CCAT_PAD; kc += 32) {
#pragma unroll
        for (int i = 0; i < NT2; ++i) {
            int t = wid + i * 8;
            if (t < T2) {
                int ot = t >> 2, mt = t & 3;
                v16h a  = load_w_frag(W, 2 * CIN_REAL, ot * 16, kc, COUT_REAL);
                v16h bf = load_b_frag(&catT[mt * 16][kc], CAT_S);
                acc2[i] = wmma_f32(a, bf, acc2[i]);
            }
        }
    }

    // ---- epilogue: bias + relu + store (padded rows compute exactly 0) ----
    {
        const int l = tid & 31;
        const int h = l >> 4, cn = l & 15;
#pragma unroll
        for (int i = 0; i < NT2; ++i) {
            int t = wid + i * 8;
            if (t < T2) {
                int ot = t >> 2, mt = t & 3;
#pragma unroll
                for (int r = 0; r < 8; ++r) {
                    int o = ot * 16 + r + 8 * h;
                    float v = acc2[i][r] + (o < COUT_REAL ? bias[o] : 0.0f);
                    if (RELU) v = v > 0.0f ? v : 0.0f;
                    Xout[xbase + (size_t)o * N_NODES + m0 + mt * 16 + cn] = (_Float16)v;
                }
            }
        }
    }
}

// ---------------- e normalization: row 0 of X0, rows 1..15 zeroed -----------
__global__ __launch_bounds__(256) void norm_kernel(const float* __restrict__ e,
                                                   _Float16* __restrict__ X0)
{
    __shared__ float rs[256];
    __shared__ float rq[256];
    const int b = blockIdx.x, tid = threadIdx.x;
    float s = 0.f, q = 0.f;
    float vals[4];
#pragma unroll
    for (int i = 0; i < 4; ++i) {
        float v = e[b * N_NODES + tid + i * 256];
        vals[i] = v; s += v; q += v * v;
    }
    rs[tid] = s; rq[tid] = q;
    __syncthreads();
    for (int off = 128; off > 0; off >>= 1) {
        if (tid < off) { rs[tid] += rs[tid + off]; rq[tid] += rq[tid + off]; }
        __syncthreads();
    }
    const float mu  = rs[0] * (1.0f / N_NODES);
    float var = rq[0] * (1.0f / N_NODES) - mu * mu;
    var = var > 0.f ? var : 0.f;
    const float inv = 1.0f / (sqrtf(var) + 1e-5f);
    const size_t xbase = (size_t)b * XROWS * N_NODES;
#pragma unroll
    for (int i = 0; i < 4; ++i)
        X0[xbase + tid + i * 256] = (_Float16)((vals[i] - mu) * inv);
    for (int idx = tid; idx < 15 * N_NODES; idx += 256)
        X0[xbase + N_NODES + idx] = (_Float16)0.0f;
}

// ---------------- head: sigmoid(mean over N of channel 0) -------------------
__global__ __launch_bounds__(256) void head_kernel(const _Float16* __restrict__ X,
                                                   float* __restrict__ out)
{
    __shared__ float rs[256];
    const int b = blockIdx.x, tid = threadIdx.x;
    const size_t xbase = (size_t)b * XROWS * N_NODES;
    float s = 0.f;
#pragma unroll
    for (int i = 0; i < 4; ++i) s += (float)X[xbase + tid + i * 256];
    rs[tid] = s;
    __syncthreads();
    for (int off = 128; off > 0; off >>= 1) {
        if (tid < off) rs[tid] += rs[tid + off];
        __syncthreads();
    }
    if (tid == 0) {
        float mean = rs[0] * (1.0f / N_NODES);
        out[b] = 1.0f / (1.0f + __expf(-mean));
    }
}

extern "C" void kernel_launch(void* const* d_in, const int* in_sizes, int n_in,
                              void* d_out, int out_size, void* d_ws, size_t ws_size,
                              hipStream_t stream)
{
    (void)in_sizes; (void)n_in; (void)out_size; (void)ws_size;
    const float* e   = (const float*)d_in[0];
    const float* phi = (const float*)d_in[1];
    const float* eta = (const float*)d_in[2];
    const float* W0  = (const float*)d_in[3];
    const float* b0  = (const float*)d_in[4];
    const float* W1  = (const float*)d_in[5];
    const float* b1  = (const float*)d_in[6];
    const float* W2  = (const float*)d_in[7];
    const float* b2  = (const float*)d_in[8];
    const float* W3  = (const float*)d_in[9];
    const float* b3  = (const float*)d_in[10];
    float* out = (float*)d_out;

    _Float16* bufA = (_Float16*)d_ws;                               // 8 MB
    _Float16* bufB = bufA + (size_t)BATCH * XROWS * N_NODES;        // 8 MB

    dim3 grid(N_NODES / MT, BATCH);   // 16 x 64 = 1024 workgroups per layer
    norm_kernel<<<BATCH, 256, 0, stream>>>(e, bufA);
    layer_kernel< 1, 16,  32, 32, 32, true ><<<grid, 256, 0, stream>>>(bufA, phi, eta, W0, b0, bufB);
    layer_kernel<32, 32,  64, 64, 64, true ><<<grid, 256, 0, stream>>>(bufB, phi, eta, W1, b1, bufA);
    layer_kernel<64, 64, 128, 64, 64, true ><<<grid, 256, 0, stream>>>(bufA, phi, eta, W2, b2, bufB);
    layer_kernel<64, 64, 128,  1, 16, false><<<grid, 256, 0, stream>>>(bufB, phi, eta, W3, b3, bufA);
    head_kernel<<<BATCH, 256, 0, stream>>>(bufA, out);
}
